// LJCH1_34548716929306
// MI455X (gfx1250) — matrix operand order repaired and checked by reading, not compiled, not run on hardware
//
#include <hip/hip_runtime.h>
#include <math.h>

// ---------------------------------------------------------------------------
// Problem constants (from reference)
// ---------------------------------------------------------------------------
#define Bn   16
#define Tn   2048
#define DC   4096
#define DD   512
#define DM   256
#define KTOT 4864      // DC + DD + DM
#define REDN 128       // fc0 output width
#define Hh   32        // GRU hidden
#define G3   96        // 3*H gates per direction
#define NGAT 192       // both directions fused
#define NEGV -1e9f

typedef __attribute__((ext_vector_type(16))) __bf16 v16bf;
typedef __attribute__((ext_vector_type(8)))  __bf16 v8bf;
typedef __attribute__((ext_vector_type(8)))  float  v8f;

// ---------------------------------------------------------------------------
// Kernel 1: scores[B*T,128] = concat(content,distort,motion) @ fc0_w^T + b
// Bandwidth-bound: A streamed once (fp32 -> bf16 in regs), W chunk staged in
// LDS as bf16, f32 accumulation via v_wmma_f32_16x16x32_bf16.
// Block: 256 threads = 8 waves; wave owns 16 rows x full N=128 (8 subtiles).
// Grid: 32768/128 = 256 blocks.
// ---------------------------------------------------------------------------
__global__ __launch_bounds__(256) void k_fc0(
    const float* __restrict__ content, const float* __restrict__ distort,
    const float* __restrict__ motion,  const float* __restrict__ fcw,
    const float* __restrict__ fcb,     __bf16* __restrict__ scores)
{
    __shared__ __bf16 sW[REDN][40];            // 128 n-rows x 32 k (pad->40)

    const int tid   = threadIdx.x;
    const int wv    = tid >> 5;
    const int lane  = tid & 31;
    const int hlane = lane & 15;
    const int m0    = blockIdx.x * 128 + wv * 16;
    const int row   = m0 + hlane;              // A row for this lane
    const int kb    = (lane < 16) ? 0 : 8;     // A-frag K base (ISA lane map)
    const int kbb   = (lane < 16) ? 0 : 8;     // B-frag K base

    v8f acc[8];
    #pragma unroll
    for (int s = 0; s < 8; ++s)
        #pragma unroll
        for (int j = 0; j < 8; ++j) acc[s][j] = 0.f;

    for (int kc = 0; kc < KTOT; kc += 32) {
        __syncthreads();                       // previous chunk fully consumed
        #pragma unroll 4
        for (int i = tid; i < REDN * 32; i += 256) {
            int n = i >> 5, k = i & 31;
            sW[n][k] = (__bf16)fcw[(size_t)n * KTOT + kc + k];
        }
        __syncthreads();

        // Segment-select the streamed A row (concat boundaries are 32-aligned)
        const float* arow;
        if (kc < DC)           arow = content + (size_t)row * DC + kc;
        else if (kc < DC + DD) arow = distort + (size_t)row * DD + (kc - DC);
        else                   arow = motion  + (size_t)row * DM + (kc - DC - DD);
        __builtin_prefetch(arow + 32, 0, 1);   // next K chunk -> global_prefetch_b8

        float4 f0 = *(const float4*)(arow + kb);
        float4 f1 = *(const float4*)(arow + kb + 4);
        float4 f2 = *(const float4*)(arow + kb + 16);
        float4 f3 = *(const float4*)(arow + kb + 20);
        float af[16] = { f0.x, f0.y, f0.z, f0.w,  f1.x, f1.y, f1.z, f1.w,
                         f2.x, f2.y, f2.z, f2.w,  f3.x, f3.y, f3.z, f3.w };
        v16bf a;
        #pragma unroll
        for (int j = 0; j < 16; ++j) a[j] = (__bf16)af[j];

        #pragma unroll
        for (int s = 0; s < 8; ++s) {
            int n = s * 16 + hlane;            // B column for this lane
            v8bf lo = *(const v8bf*)&sW[n][kbb];
            v8bf hi = *(const v8bf*)&sW[n][kbb + 16];
            v16bf bfrag;
            #pragma unroll
            for (int j = 0; j < 8; ++j) { bfrag[j] = lo[j]; bfrag[j + 8] = hi[j]; }
            acc[s] = __builtin_amdgcn_wmma_f32_16x16x32_bf16(
                false, a, false, bfrag, (short)0, acc[s], false, false);
        }
    }

    // C/D layout: VGPR r, lanes 0-15 -> M=r, lanes 16-31 -> M=8+r; N = lane&15
    const int rbase = m0 + ((lane >= 16) ? 8 : 0);
    #pragma unroll
    for (int s = 0; s < 8; ++s) {
        int col = s * 16 + hlane;
        float bias = fcb[col];
        #pragma unroll
        for (int r = 0; r < 8; ++r)
            scores[(size_t)(rbase + r) * REDN + col] = (__bf16)(acc[s][r] + bias);
    }
}

// ---------------------------------------------------------------------------
// Kernel 2: gi[B*T,192] = scores_bf16 @ [wi_f; wi_b]^T + [bi_f; bi_b]
// K=128 (4 chunks), N=192 (12 subtiles/wave). Weights staged once in LDS.
// ---------------------------------------------------------------------------
__global__ __launch_bounds__(256) void k_gates(
    const __bf16* __restrict__ scores,
    const float* __restrict__ wiF, const float* __restrict__ wiB,
    const float* __restrict__ biF, const float* __restrict__ biB,
    float* __restrict__ gi)
{
    __shared__ __bf16 sWi[NGAT][136];          // 192 x 128 k (pad->136)

    const int tid   = threadIdx.x;
    const int wv    = tid >> 5;
    const int lane  = tid & 31;
    const int hlane = lane & 15;
    const int m0    = blockIdx.x * 128 + wv * 16;
    const int row   = m0 + hlane;
    const int kb    = (lane < 16) ? 0 : 8;
    const int kbb   = (lane < 16) ? 0 : 8;

    for (int i = tid; i < NGAT * REDN; i += 256) {
        int n = i >> 7, k = i & 127;
        float v = (n < G3) ? wiF[n * REDN + k] : wiB[(n - G3) * REDN + k];
        sWi[n][k] = (__bf16)v;
    }
    __syncthreads();

    v8f acc[12];
    #pragma unroll
    for (int s = 0; s < 12; ++s)
        #pragma unroll
        for (int j = 0; j < 8; ++j) acc[s][j] = 0.f;

    const __bf16* arow = scores + (size_t)row * REDN;
    #pragma unroll
    for (int kc = 0; kc < REDN; kc += 32) {
        v8bf a0 = *(const v8bf*)(arow + kc + kb);
        v8bf a1 = *(const v8bf*)(arow + kc + kb + 16);
        v16bf a;
        #pragma unroll
        for (int j = 0; j < 8; ++j) { a[j] = a0[j]; a[j + 8] = a1[j]; }

        #pragma unroll
        for (int s = 0; s < 12; ++s) {
            int n = s * 16 + hlane;
            v8bf lo = *(const v8bf*)&sWi[n][kc + kbb];
            v8bf hi = *(const v8bf*)&sWi[n][kc + kbb + 16];
            v16bf bfrag;
            #pragma unroll
            for (int j = 0; j < 8; ++j) { bfrag[j] = lo[j]; bfrag[j + 8] = hi[j]; }
            acc[s] = __builtin_amdgcn_wmma_f32_16x16x32_bf16(
                false, a, false, bfrag, (short)0, acc[s], false, false);
        }
    }

    const int rbase = m0 + ((lane >= 16) ? 8 : 0);
    #pragma unroll
    for (int s = 0; s < 12; ++s) {
        int col = s * 16 + hlane;
        float bias = (col < G3) ? biF[col] : biB[col - G3];
        #pragma unroll
        for (int r = 0; r < 8; ++r)
            gi[(size_t)(rbase + r) * NGAT + col] = acc[s][r] + bias;
    }
}

// ---------------------------------------------------------------------------
// Kernel 3: sequential bidirectional GRU scan, q-projection fused.
// grid = 2 (direction); block = 512 = 16 batch-waves x 32 hidden lanes.
// State kept in LDS; q contribution reduced with wave32 shuffles.
// ---------------------------------------------------------------------------
__global__ __launch_bounds__(512) void k_scan(
    const float* __restrict__ gi,
    const float* __restrict__ whF, const float* __restrict__ whB,
    const float* __restrict__ bhF, const float* __restrict__ bhB,
    const float* __restrict__ qw,
    float* __restrict__ qf, float* __restrict__ qb)
{
    __shared__ float sWh[G3 * Hh];             // 96x32 recurrent weights
    __shared__ float sH[Bn][Hh + 1];           // hidden state, padded

    const int dir  = blockIdx.x;
    const float* wh  = dir ? whB : whF;
    const float* bhv = dir ? bhB : bhF;
    float* qout      = dir ? qb  : qf;

    const int tid = threadIdx.x;
    const int b   = tid >> 5;
    const int h   = tid & 31;

    for (int i = tid; i < G3 * Hh; i += 512) sWh[i] = wh[i];
    sH[b][h] = 0.f;
    float hreg = 0.f;
    const float qwv = qw[dir * Hh + h];
    const float br  = bhv[h], bz = bhv[Hh + h], bn2 = bhv[2 * Hh + h];
    const float* grow = gi + (size_t)b * Tn * NGAT + dir * G3;
    __syncthreads();

    for (int s = 0; s < Tn; ++s) {
        const int t = dir ? (Tn - 1 - s) : s;
        const float* g = grow + (size_t)t * NGAT;

        float hr = br, hz = bz, hn = bn2;
        #pragma unroll 8
        for (int k = 0; k < Hh; ++k) {
            float hk = sH[b][k];
            hr += sWh[h * Hh + k] * hk;
            hz += sWh[(Hh + h) * Hh + k] * hk;
            hn += sWh[(2 * Hh + h) * Hh + k] * hk;
        }
        float r = 1.f / (1.f + __expf(-(g[h] + hr)));
        float z = 1.f / (1.f + __expf(-(g[Hh + h] + hz)));
        float n = tanhf(g[2 * Hh + h] + r * hn);
        float hnew = (1.f - z) * n + z * hreg;
        hreg = hnew;

        float pq = hnew * qwv;                 // fused q = h . q_w slice
        #pragma unroll
        for (int o = 16; o; o >>= 1) pq += __shfl_xor(pq, o, 32);
        if (h == 0) qout[b * Tn + t] = pq;

        __syncthreads();
        sH[b][h] = hnew;
        __syncthreads();
    }
}

// ---------------------------------------------------------------------------
// Kernel 4: masked 'same' conv (k=3,5,7) + softmax-weighted pooling per row.
// grid = B; block = 256; entire signal in LDS; two-pass stable softmax.
// ---------------------------------------------------------------------------
__device__ __forceinline__ float wave_sum(float v) {
    #pragma unroll
    for (int o = 16; o; o >>= 1) v += __shfl_xor(v, o, 32);
    return v;
}
__device__ __forceinline__ float wave_max(float v) {
    #pragma unroll
    for (int o = 16; o; o >>= 1) v = fmaxf(v, __shfl_xor(v, o, 32));
    return v;
}

__global__ __launch_bounds__(256) void k_pool(
    const float* __restrict__ qf, const float* __restrict__ qb,
    const float* __restrict__ qbias, const int* __restrict__ inputLength,
    const float* __restrict__ w1, const float* __restrict__ w2,
    const float* __restrict__ w3, float* __restrict__ out)
{
    __shared__ float sq[Tn];
    __shared__ float sc[Tn];
    __shared__ float sred[8];

    const int b = blockIdx.x, tid = threadIdx.x;
    const int wv = tid >> 5, lane = tid & 31;
    const int len = inputLength[b] - 3;        // inputLength - 2*(TI/2) - 1
    const float qb0 = qbias[0];

    for (int t = tid; t < Tn; t += 256) {
        float v = qf[b * Tn + t] + qb[b * Tn + t] + qb0;
        sq[t] = (t < len) ? v : 0.f;
    }
    __syncthreads();

    const float* wks[3] = { w1, w2, w3 };
    const int    ksz[3] = { 3, 5, 7 };
    float total = 0.f;

    for (int c = 0; c < 3; ++c) {
        const float* wk = wks[c];
        const int K = ksz[c], p = K >> 1;

        for (int t = tid; t < Tn; t += 256) {
            float a = 0.f;
            for (int j = 0; j < K; ++j) {
                int idx = t + j - p;
                if (idx >= 0 && idx < Tn) a += wk[j] * sq[idx];
            }
            sc[t] = (t < len) ? a : NEGV;
        }
        __syncthreads();

        float m = -INFINITY;
        for (int t = tid; t < Tn; t += 256) m = fmaxf(m, sc[t]);
        m = wave_max(m);
        if (lane == 0) sred[wv] = m;
        __syncthreads();
        float bm = sred[0];
        #pragma unroll
        for (int i = 1; i < 8; ++i) bm = fmaxf(bm, sred[i]);
        __syncthreads();

        float se = 0.f, sw = 0.f;
        for (int t = tid; t < Tn; t += 256) {
            float e = __expf(sc[t] - bm);
            se += e;
            sw += e * sq[t];
        }
        se = wave_sum(se);
        sw = wave_sum(sw);
        if (lane == 0) sred[wv] = se;
        __syncthreads();
        float bse = 0.f;
        #pragma unroll
        for (int i = 0; i < 8; ++i) bse += sred[i];
        __syncthreads();
        if (lane == 0) sred[wv] = sw;
        __syncthreads();
        float bsw = 0.f;
        #pragma unroll
        for (int i = 0; i < 8; ++i) bsw += sred[i];
        __syncthreads();

        total += bsw / bse;
    }

    if (tid == 0) out[b] = total / 3.f;
}

// ---------------------------------------------------------------------------
// Host-side launcher
// ---------------------------------------------------------------------------
extern "C" void kernel_launch(void* const* d_in, const int* in_sizes, int n_in,
                              void* d_out, int out_size, void* d_ws, size_t ws_size,
                              hipStream_t stream)
{
    const float* motion  = (const float*)d_in[0];
    const float* content = (const float*)d_in[1];
    const float* distort = (const float*)d_in[2];
    const int*   inLen   = (const int*)  d_in[3];
    const float* fcw     = (const float*)d_in[4];
    const float* fcb     = (const float*)d_in[5];
    const float* wiF     = (const float*)d_in[6];
    const float* whF     = (const float*)d_in[7];
    const float* biF     = (const float*)d_in[8];
    const float* bhF     = (const float*)d_in[9];
    const float* wiB     = (const float*)d_in[10];
    const float* whB     = (const float*)d_in[11];
    const float* biB     = (const float*)d_in[12];
    const float* bhB     = (const float*)d_in[13];
    const float* qw      = (const float*)d_in[14];
    const float* qbias   = (const float*)d_in[15];
    const float* w1      = (const float*)d_in[16];
    const float* w2      = (const float*)d_in[17];
    const float* w3      = (const float*)d_in[18];
    float* out = (float*)d_out;

    const size_t M = (size_t)Bn * Tn;          // 32768 rows

    // Workspace layout (~33.7 MB total)
    char* ws = (char*)d_ws;
    __bf16* scores = (__bf16*)ws;  ws += M * REDN * sizeof(__bf16);   // 8 MB
    float*  gi     = (float*)ws;   ws += M * NGAT * sizeof(float);    // 25 MB
    float*  qfv    = (float*)ws;   ws += M * sizeof(float);           // 128 KB
    float*  qbv    = (float*)ws;

    k_fc0  <<<(int)(M / 128), 256, 0, stream>>>(content, distort, motion, fcw, fcb, scores);
    k_gates<<<(int)(M / 128), 256, 0, stream>>>(scores, wiF, wiB, biF, biB, gi);
    k_scan <<<2, 512, 0, stream>>>(gi, whF, whB, bhF, bhB, qw, qfv, qbv);
    k_pool <<<Bn, 256, 0, stream>>>(qfv, qbv, qbias, inLen, w1, w2, w3, out);
}
